// SelfAttention_11630771437791
// MI455X (gfx1250) — matrix operand order repaired and checked by reading, not compiled, not run on hardware
//
#include <hip/hip_runtime.h>
#include <hip/hip_bf16.h>

// ---------------- CDNA5 wave32 / WMMA bf16 types ----------------
typedef __attribute__((ext_vector_type(16))) __bf16 v16bf;
typedef __attribute__((ext_vector_type(8)))  float  v8f;

union FragBF {
  v16bf v;
  unsigned int u[8];
  unsigned short h[16];
};

#define BATCH 16
#define CCH   256
#define NSP   4096
#define NHEAD 4
#define HDIM  64
#define SPLITK 8          // split-K factor for the context GEMM (K = 4096 -> 8 x 512)

__device__ __forceinline__ unsigned short f2bf(float f) {
  unsigned int u = __float_as_uint(f);
  u += 0x7FFFu + ((u >> 16) & 1u);          // round-to-nearest-even
  return (unsigned short)(u >> 16);
}
__device__ __forceinline__ float bf2f(unsigned short h) {
  return __uint_as_float(((unsigned int)h) << 16);
}

// A fragment: 16(M) x 32(K) bf16 from row-major [M][ld] memory.
// ISA 7.12.2: lanes 0-15 -> M=lane, K pairs {2p,2p+1} for p<4, {16+2p..} for p>=4;
// lanes 16-31 same M, K base +8.
__device__ __forceinline__ v16bf load_frag_a(const unsigned short* p, int ld, int lane) {
  FragBF f;
  const unsigned short* row = p + (size_t)(lane & 15) * ld;
  int kb = (lane >> 4) << 3;
#pragma unroll
  for (int i = 0; i < 4; ++i) {
    f.u[i]     = *(const unsigned int*)(row + kb + 2 * i);
    f.u[i + 4] = *(const unsigned int*)(row + 16 + kb + 2 * i);
  }
  return f.v;
}

// B fragment: 32(K) x 16(N) bf16; memory is N-major [N][ld] with K contiguous.
// Lanes 0-15: N=lane, K=0..7,16..23; lanes 16-31: same N, K=8..15,24..31 (matches A).
__device__ __forceinline__ v16bf load_frag_b(const unsigned short* p, int ld, int lane) {
  FragBF f;
  const unsigned short* row = p + (size_t)(lane & 15) * ld;
  int kb = (lane >> 4) << 3;
#pragma unroll
  for (int i = 0; i < 4; ++i) {
    f.u[i]     = *(const unsigned int*)(row + kb + 2 * i);
    f.u[i + 4] = *(const unsigned int*)(row + 16 + kb + 2 * i);
  }
  return f.v;
}

#define WMMA_BF16(a, b, c) \
  __builtin_amdgcn_wmma_f32_16x16x32_bf16(false, (a), false, (b), (short)0, (c), false, false)

// ---------------- 1) GroupNorm statistics ----------------
__global__ void __launch_bounds__(256) gn_stats_kernel(const float* __restrict__ x,
                                                       float* __restrict__ stats) {
  int bg = blockIdx.x;                 // b*4 + g
  int b = bg >> 2, g = bg & 3;
  const float* base = x + ((size_t)(b * CCH + g * 64)) * NSP;
  float s = 0.f, ss = 0.f;
  for (int i = threadIdx.x; i < 64 * NSP; i += 256) {
    float v = base[i];
    s += v; ss += v * v;
  }
  __shared__ float sh[512];
  sh[threadIdx.x] = s; sh[256 + threadIdx.x] = ss;
  __syncthreads();
  for (int off = 128; off > 0; off >>= 1) {
    if (threadIdx.x < off) {
      sh[threadIdx.x]       += sh[threadIdx.x + off];
      sh[256 + threadIdx.x] += sh[256 + threadIdx.x + off];
    }
    __syncthreads();
  }
  if (threadIdx.x == 0) {
    const float inv = 1.f / (64.f * NSP);
    float mean = sh[0] * inv;
    float var  = sh[256] * inv - mean * mean;
    stats[bg * 2]     = mean;
    stats[bg * 2 + 1] = rsqrtf(var + 1e-5f);
  }
}

// ---------------- 2) normalize + affine, write normT[b][n][c] bf16 ----------------
__global__ void __launch_bounds__(256) gn_apply_kernel(const float* __restrict__ x,
                                                       const float* __restrict__ gn_w,
                                                       const float* __restrict__ gn_b,
                                                       const float* __restrict__ stats,
                                                       unsigned short* __restrict__ normT) {
  size_t i = (size_t)blockIdx.x * 256 + threadIdx.x;   // exactly B*N*C threads
  int c = (int)(i & 255);
  int n = (int)((i >> 8) & 4095);
  int b = (int)(i >> 20);
  int bg = b * 4 + (c >> 6);
  float mu = stats[bg * 2], rs = stats[bg * 2 + 1];
  float v = x[((size_t)(b * CCH + c)) * NSP + n];
  normT[i] = f2bf((v - mu) * rs * gn_w[c] + gn_b[c]);
}

// ---------------- weight f32 -> bf16 ----------------
__global__ void cvt_bf16_kernel(const float* __restrict__ in, unsigned short* __restrict__ out,
                                int n) {
  int i = blockIdx.x * 256 + threadIdx.x;
  if (i < n) out[i] = f2bf(in[i]);
}

// ---------------- 3) QKV GEMM: qkv[o,n] = sum_c Wqkv[o,c] * normT[n,c] ----------------
// grid (N/128, 768/16, B); wave = one 16x16 tile; K = 256 (8 WMMA steps).
__global__ void __launch_bounds__(256) qkv_gemm_kernel(const unsigned short* __restrict__ wq,
                                                       const unsigned short* __restrict__ normT,
                                                       unsigned short* __restrict__ qT,
                                                       unsigned short* __restrict__ kbuf,
                                                       unsigned short* __restrict__ vbuf) {
  int lane = threadIdx.x & 31, wave = threadIdx.x >> 5;
  int b  = blockIdx.z;
  int o0 = blockIdx.y * 16;
  int n0 = blockIdx.x * 128 + wave * 16;
  const unsigned short* ap = wq + (size_t)o0 * CCH;
  const unsigned short* bp = normT + ((size_t)b * NSP + n0) * CCH;
  v8f acc = {0.f, 0.f, 0.f, 0.f, 0.f, 0.f, 0.f, 0.f};
#pragma unroll
  for (int kt = 0; kt < 8; ++kt) {
    v16bf a  = load_frag_a(ap + kt * 32, CCH, lane);
    v16bf bb = load_frag_b(bp + kt * 32, CCH, lane);
    acc = WMMA_BF16(a, bb, acc);
  }
  int col = lane & 15, rbase = (lane >> 4) << 3;
  int n = n0 + col;
#pragma unroll
  for (int r = 0; r < 8; ++r) {
    int o = o0 + rbase + r;
    unsigned short hv = f2bf(acc[r]);
    int which = o >> 8, c = o & 255, h = c >> 6, d = c & 63;
    size_t bh = (size_t)(b * NHEAD + h);
    if (which == 0)      qT[(bh * NSP + n) * HDIM + d]  = hv;  // q transposed [n][d]
    else if (which == 1) kbuf[(bh * HDIM + d) * NSP + n] = hv; // k [d][n]
    else                 vbuf[(bh * HDIM + d) * NSP + n] = hv; // v [d][n]
  }
}

// ---------------- 4) softmax over n for each k row (4096 rows x 4096) ----------------
__global__ void __launch_bounds__(256) softmax_kernel(unsigned short* __restrict__ kbuf) {
  unsigned short* p = kbuf + (size_t)blockIdx.x * NSP;
  float vals[16];
  float mx = -3.4e38f;
#pragma unroll
  for (int j = 0; j < 16; ++j) {
    float v = bf2f(p[threadIdx.x + j * 256]);
    vals[j] = v;
    mx = fmaxf(mx, v);
  }
  __shared__ float sh[256];
  sh[threadIdx.x] = mx; __syncthreads();
  for (int off = 128; off > 0; off >>= 1) {
    if (threadIdx.x < off) sh[threadIdx.x] = fmaxf(sh[threadIdx.x], sh[threadIdx.x + off]);
    __syncthreads();
  }
  mx = sh[0]; __syncthreads();
  float s = 0.f;
#pragma unroll
  for (int j = 0; j < 16; ++j) { vals[j] = __expf(vals[j] - mx); s += vals[j]; }
  sh[threadIdx.x] = s; __syncthreads();
  for (int off = 128; off > 0; off >>= 1) {
    if (threadIdx.x < off) sh[threadIdx.x] += sh[threadIdx.x + off];
    __syncthreads();
  }
  float inv = 1.f / sh[0];
#pragma unroll
  for (int j = 0; j < 16; ++j) p[threadIdx.x + j * 256] = f2bf(vals[j] * inv);
}

// ---- 5) context[d,e] = sum_n k[d,n]*v[e,n], split-K partials (f32) ----
// grid (B*H, SPLITK); 8 waves x 2 tiles cover 64x64; each split reduces 512 n.
__global__ void __launch_bounds__(256) ctx_gemm_kernel(const unsigned short* __restrict__ kbuf,
                                                       const unsigned short* __restrict__ vbuf,
                                                       float* __restrict__ ctx_part) {
  int lane = threadIdx.x & 31, wave = threadIdx.x >> 5;
  size_t bh = blockIdx.x;
  int split = blockIdx.y;
  int kt0 = split * (NSP / SPLITK / 32);           // 16 kt iterations per split
  int e0  = (wave & 3) * 16;
  int d0a = (wave >> 2) * 16;
  int d0b = d0a + 32;
  const unsigned short* ka = kbuf + (bh * HDIM + d0a) * NSP;
  const unsigned short* kb = kbuf + (bh * HDIM + d0b) * NSP;
  const unsigned short* vp = vbuf + (bh * HDIM + e0) * NSP;
  v8f acc0 = {0.f, 0.f, 0.f, 0.f, 0.f, 0.f, 0.f, 0.f};
  v8f acc1 = acc0;
#pragma unroll 4
  for (int kk = 0; kk < NSP / SPLITK / 32; ++kk) {
    int kt = kt0 + kk;
    __builtin_prefetch(vp + kt * 32 + 256, 0, 1);   // global_prefetch_b8, ahead in row
    __builtin_prefetch(ka + kt * 32 + 256, 0, 1);
    v16bf bfr = load_frag_b(vp + kt * 32, NSP, lane);  // B tile shared by both d-tiles
    v16bf a0  = load_frag_a(ka + kt * 32, NSP, lane);
    v16bf a1  = load_frag_a(kb + kt * 32, NSP, lane);
    acc0 = WMMA_BF16(a0, bfr, acc0);
    acc1 = WMMA_BF16(a1, bfr, acc1);
  }
  int col = lane & 15, rbase = (lane >> 4) << 3;
  float* outp = ctx_part + (bh * SPLITK + split) * (HDIM * HDIM);
#pragma unroll
  for (int r = 0; r < 8; ++r) {
    outp[(size_t)(e0 + col) * HDIM + d0a + rbase + r] = acc0[r];   // ctxT layout [e][d]
    outp[(size_t)(e0 + col) * HDIM + d0b + rbase + r] = acc1[r];
  }
}

// ---- 5b) reduce split-K partials -> ctxT[e][d] bf16 ----
__global__ void __launch_bounds__(256) ctx_reduce_kernel(const float* __restrict__ ctx_part,
                                                         unsigned short* __restrict__ ctxT) {
  size_t i = (size_t)blockIdx.x * 256 + threadIdx.x;   // B*H*64*64 elements
  size_t bh = i >> 12;
  size_t idx = i & 4095;
  const float* p = ctx_part + bh * SPLITK * (HDIM * HDIM) + idx;
  float s = 0.f;
#pragma unroll
  for (int k = 0; k < SPLITK; ++k) s += p[(size_t)k * HDIM * HDIM];
  ctxT[i] = f2bf(s);
}

// ---------------- 6) out2[e,n] = sum_d ctxT[e,d]*qT[n,d]; store attnT[b][n][h*64+e] ---
// grid (N/128, 4, B*H); K = 64 (2 WMMA steps).
__global__ void __launch_bounds__(256) out2_gemm_kernel(const unsigned short* __restrict__ ctxT,
                                                        const unsigned short* __restrict__ qT,
                                                        unsigned short* __restrict__ attnT) {
  int lane = threadIdx.x & 31, wave = threadIdx.x >> 5;
  int bh = blockIdx.z, b = bh >> 2, h = bh & 3;
  int e0 = blockIdx.y * 16;
  int n0 = blockIdx.x * 128 + wave * 16;
  const unsigned short* ap = ctxT + (size_t)bh * (HDIM * HDIM) + e0 * HDIM;
  const unsigned short* bp = qT + ((size_t)bh * NSP + n0) * HDIM;
  v8f acc = {0.f, 0.f, 0.f, 0.f, 0.f, 0.f, 0.f, 0.f};
#pragma unroll
  for (int kt = 0; kt < 2; ++kt) {
    v16bf a  = load_frag_a(ap + kt * 32, HDIM, lane);
    v16bf bb = load_frag_b(bp + kt * 32, HDIM, lane);
    acc = WMMA_BF16(a, bb, acc);
  }
  int col = lane & 15, rbase = (lane >> 4) << 3;
  int n = n0 + col;
#pragma unroll
  for (int r = 0; r < 8; ++r) {
    int e = e0 + rbase + r;
    attnT[((size_t)b * NSP + n) * CCH + h * HDIM + e] = f2bf(acc[r]);
  }
}

// ------- 7) y[o,n] = sum_c Wout[o,c]*attnT[n,c] + b_out[o] + x[o,n] (f32 out) -------
// grid (N/128, 16, B); K = 256 (8 WMMA steps).
__global__ void __launch_bounds__(256) out_gemm_kernel(const unsigned short* __restrict__ wob,
                                                       const unsigned short* __restrict__ attnT,
                                                       const float* __restrict__ b_out,
                                                       const float* __restrict__ x,
                                                       float* __restrict__ y) {
  int lane = threadIdx.x & 31, wave = threadIdx.x >> 5;
  int b  = blockIdx.z;
  int o0 = blockIdx.y * 16;
  int n0 = blockIdx.x * 128 + wave * 16;
  const unsigned short* ap = wob + (size_t)o0 * CCH;
  const unsigned short* bp = attnT + ((size_t)b * NSP + n0) * CCH;
  v8f acc = {0.f, 0.f, 0.f, 0.f, 0.f, 0.f, 0.f, 0.f};
#pragma unroll
  for (int kt = 0; kt < 8; ++kt) {
    v16bf a  = load_frag_a(ap + kt * 32, CCH, lane);
    v16bf bb = load_frag_b(bp + kt * 32, CCH, lane);
    acc = WMMA_BF16(a, bb, acc);
  }
  int col = lane & 15, rbase = (lane >> 4) << 3;
  int n = n0 + col;
#pragma unroll
  for (int r = 0; r < 8; ++r) {
    int o = o0 + rbase + r;
    size_t idx = ((size_t)(b * CCH + o)) * NSP + n;
    y[idx] = acc[r] + b_out[o] + x[idx];
  }
}

// ---------------- host side ----------------
extern "C" void kernel_launch(void* const* d_in, const int* in_sizes, int n_in,
                              void* d_out, int out_size, void* d_ws, size_t ws_size,
                              hipStream_t stream) {
  const float* x     = (const float*)d_in[0];
  const float* gn_w  = (const float*)d_in[1];
  const float* gn_b  = (const float*)d_in[2];
  const float* w_qkv = (const float*)d_in[3];
  const float* w_out = (const float*)d_in[4];
  const float* b_out = (const float*)d_in[5];
  float* out = (float*)d_out;

  char* ws = (char*)d_ws;
  size_t off = 0;
  auto carve = [&](size_t bytes) -> void* {
    void* p = ws + off;
    off += (bytes + 255) & ~(size_t)255;
    return p;
  };
  float*          stats    = (float*)carve(BATCH * 4 * 2 * sizeof(float));
  unsigned short* wqkv_b   = (unsigned short*)carve((size_t)3 * CCH * CCH * 2);
  unsigned short* wout_b   = (unsigned short*)carve((size_t)CCH * CCH * 2);
  unsigned short* normT    = (unsigned short*)carve((size_t)BATCH * NSP * CCH * 2);
  unsigned short* qT       = (unsigned short*)carve((size_t)BATCH * NHEAD * NSP * HDIM * 2);
  unsigned short* kbuf     = (unsigned short*)carve((size_t)BATCH * NHEAD * HDIM * NSP * 2);
  unsigned short* vbuf     = (unsigned short*)carve((size_t)BATCH * NHEAD * HDIM * NSP * 2);
  float*          ctx_part = (float*)carve((size_t)BATCH * NHEAD * SPLITK * HDIM * HDIM * 4);
  unsigned short* ctxT     = (unsigned short*)carve((size_t)BATCH * NHEAD * HDIM * HDIM * 2);
  unsigned short* attnT    = (unsigned short*)carve((size_t)BATCH * NSP * CCH * 2);
  (void)ws_size; (void)in_sizes; (void)n_in; (void)out_size;

  gn_stats_kernel<<<BATCH * 4, 256, 0, stream>>>(x, stats);
  cvt_bf16_kernel<<<(3 * CCH * CCH + 255) / 256, 256, 0, stream>>>(w_qkv, wqkv_b, 3 * CCH * CCH);
  cvt_bf16_kernel<<<(CCH * CCH + 255) / 256, 256, 0, stream>>>(w_out, wout_b, CCH * CCH);
  gn_apply_kernel<<<(BATCH * NSP * CCH) / 256, 256, 0, stream>>>(x, gn_w, gn_b, stats, normT);
  qkv_gemm_kernel<<<dim3(NSP / 128, 48, BATCH), 256, 0, stream>>>(wqkv_b, normT, qT, kbuf, vbuf);
  softmax_kernel<<<BATCH * NHEAD * HDIM, 256, 0, stream>>>(kbuf);
  ctx_gemm_kernel<<<dim3(BATCH * NHEAD, SPLITK), 256, 0, stream>>>(kbuf, vbuf, ctx_part);
  ctx_reduce_kernel<<<(BATCH * NHEAD * HDIM * HDIM) / 256, 256, 0, stream>>>(ctx_part, ctxT);
  out2_gemm_kernel<<<dim3(NSP / 128, 4, BATCH * NHEAD), 256, 0, stream>>>(ctxT, qT, attnT);
  out_gemm_kernel<<<dim3(NSP / 128, 16, BATCH), 256, 0, stream>>>(wout_b, attnT, b_out, x, out);
}
